// HTNet_6305011990584
// MI455X (gfx1250) — compile-verified, matrix operand order checked by reading, not run on hardware
//
#include <hip/hip_runtime.h>
#include <hip/hip_bf16.h>

typedef __attribute__((ext_vector_type(16))) _Float16 v16h;
typedef __attribute__((ext_vector_type(8)))  float    v8f;

#define B_    1024
#define NR_   6
#define MPR_  12
#define TOK_  (B_*NR_*MPR_)   // 73728
#define SEQ_  (B_*NR_)        // 6144
#define DIM_  256
#define PD_   147
#define PDP_  160             // 147 padded to multiple of 32

__device__ __forceinline__ float wave_sum(float v) {
#pragma unroll
  for (int m = 16; m > 0; m >>= 1) v += __shfl_xor(v, m, 32);
  return v;
}

// ---- WMMA fragment loaders (A 16x32 f16 layout: lanes0-15 K{0..7,16..23}, lanes16-31 K{8..15,24..31}) ----
__device__ __forceinline__ v16h frag_f16(const _Float16* __restrict__ p) {
  v16h f;
#pragma unroll
  for (int i = 0; i < 8; i++) f[i] = p[i];
#pragma unroll
  for (int i = 0; i < 8; i++) f[8 + i] = p[16 + i];
  return f;
}
__device__ __forceinline__ v16h frag_f32(const float* __restrict__ p) {
  v16h f;
#pragma unroll
  for (int i = 0; i < 8; i++) f[i] = (_Float16)p[i];
#pragma unroll
  for (int i = 0; i < 8; i++) f[8 + i] = (_Float16)p[16 + i];
  return f;
}

// ---- Generic WMMA GEMM: C = epi(A @ Bt^T + bias [+ resid]) ----
// Each wave computes a 32(M) x 64(N) tile. Per 32-K step: stage 2 A-frags + 4 B-frags
// (12 b128 loads issued as a batch), then burst 8 WMMAs so the in-order LOADcnt lets
// later fragment loads overlap earlier WMMAs (partial s_wait_loadcnt instead of 0).
// ASRC: 0 = A f16, 1 = A f32 (converted on load)
// MODE: 0 = f16 out (+opt bias), 1 = f16 out GELU(bias+..), 2 = f32 out (+opt bias, +opt resid),
//       3 = f32 out with per-region bias (bias is [6,256], region=(m/12)%6)
template <int ASRC, int MODE>
__global__ void __launch_bounds__(256) gemm_k(
    const void* __restrict__ Av, const _Float16* __restrict__ Bt,
    const float* __restrict__ bias, const float* __restrict__ resid,
    void* __restrict__ Cv, int M, int N, int K) {
  const int lane  = threadIdx.x & 31;
  const int wv    = threadIdx.x >> 5;
  const int mTile = (blockIdx.y * 8 + wv) * 2;        // two 16-row tiles per wave
  const int row0  = mTile * 16 + (lane & 15);
  const int half8 = (lane >> 4) * 8;
  const int nBase = blockIdx.x * 64;
  const int nCol  = lane & 15;
  const _Float16* A16 = (const _Float16*)Av;
  const float*    A32 = (const float*)Av;

  v8f acc0[4] = {};
  v8f acc1[4] = {};
  for (int k0 = 0; k0 < K; k0 += 32) {
    v16h a0, a1;
    if constexpr (ASRC == 0) {
      a0 = frag_f16(A16 + (size_t)row0 * K + k0 + half8);
      a1 = frag_f16(A16 + (size_t)(row0 + 16) * K + k0 + half8);
    } else {
      a0 = frag_f32(A32 + (size_t)row0 * K + k0 + half8);
      a1 = frag_f32(A32 + (size_t)(row0 + 16) * K + k0 + half8);
    }
    v16h bf[4];
#pragma unroll
    for (int t = 0; t < 4; t++)
      bf[t] = frag_f16(Bt + (size_t)(nBase + t * 16 + nCol) * K + k0 + half8);
#pragma unroll
    for (int t = 0; t < 4; t++) {
      acc0[t] = __builtin_amdgcn_wmma_f32_16x16x32_f16(false, a0, false, bf[t],
                                                       (short)0, acc0[t], false, false);
      acc1[t] = __builtin_amdgcn_wmma_f32_16x16x32_f16(false, a1, false, bf[t],
                                                       (short)0, acc1[t], false, false);
    }
  }
  // C layout: m = tile*16 + v + (lane>=16 ? 8 : 0), n = nBase + t*16 + (lane&15)
#pragma unroll
  for (int r = 0; r < 2; r++) {
    const int mBase = (mTile + r) * 16 + half8;
    const v8f* acc = r ? acc1 : acc0;
#pragma unroll
    for (int t = 0; t < 4; t++) {
      const int n = nBase + t * 16 + nCol;
      float bv = 0.f;
      if constexpr (MODE != 3) { if (bias) bv = bias[n]; }
#pragma unroll
      for (int v = 0; v < 8; v++) {
        const int m = mBase + v;
        const size_t ci = (size_t)m * N + n;
        float val = acc[t][v];
        if constexpr (MODE == 3) {
          const int reg = (m / 12) % 6;
          ((float*)Cv)[ci] = val + bias[reg * 256 + n];
        } else if constexpr (MODE == 2) {
          val += bv;
          if (resid) val += resid[ci];
          ((float*)Cv)[ci] = val;
        } else if constexpr (MODE == 1) {
          val += bv;
          val = 0.5f * val * (1.f + erff(val * 0.70710678f));
          ((_Float16*)Cv)[ci] = (_Float16)val;
        } else {
          ((_Float16*)Cv)[ci] = (_Float16)(val + bv);
        }
      }
    }
  }
}

// ---- LayerNorm (256 wide), f32 in -> f16 out, one wave per row ----
__global__ void __launch_bounds__(256) ln_k(const float* __restrict__ x,
                                            const float* __restrict__ g,
                                            const float* __restrict__ bb,
                                            _Float16* __restrict__ y, int M) {
  const int lane = threadIdx.x & 31, wv = threadIdx.x >> 5;
  const int row = blockIdx.x * 8 + wv;
  if (row >= M) return;
  const float* xr = x + (size_t)row * 256;
  float v[8], s = 0.f, ss = 0.f;
#pragma unroll
  for (int i = 0; i < 8; i++) { float u = xr[lane * 8 + i]; v[i] = u; s += u; ss += u * u; }
  s = wave_sum(s); ss = wave_sum(ss);
  const float m = s * (1.f / 256.f);
  const float r = rsqrtf(ss * (1.f / 256.f) - m * m + 1e-5f);
  _Float16* yr = y + (size_t)row * 256;
#pragma unroll
  for (int i = 0; i < 8; i++) {
    const int d = lane * 8 + i;
    yr[d] = (_Float16)((v[i] - m) * r * g[d] + bb[d]);
  }
}

// ---- Attention: one wave per (sequence, head); n=12, d=64, additive spatial bias from keypoints ----
__global__ void __launch_bounds__(256) attn_k(const _Float16* __restrict__ qkv,
                                              const float* __restrict__ kp,
                                              _Float16* __restrict__ o) {
  const int lane = threadIdx.x & 31;
  const int gw = (blockIdx.x * 256 + threadIdx.x) >> 5;
  const int s = gw >> 2, h = gw & 3;
  __shared__ float sc_s[8][144];
  float* sc = sc_s[threadIdx.x >> 5];
  const _Float16* q = qkv + (size_t)s * 12 * 768 + h * 64;
  const _Float16* k = q + 256;
  const _Float16* v = q + 512;
  const float* kps = kp + (size_t)s * 24;
  for (int e = lane; e < 144; e += 32) {
    const int i = e / 12, j = e % 12;
    float d = 0.f;
#pragma unroll
    for (int t = 0; t < 64; t++) d += (float)q[i * 768 + t] * (float)k[j * 768 + t];
    const float xi = kps[2 * i], yi = kps[2 * i + 1], xj = kps[2 * j], yj = kps[2 * j + 1];
    const float dx = xi - xj, dy = yi - yj, d2 = dx * dx + dy * dy;
    const float dist = (d2 > 0.f) ? sqrtf(d2) : 0.f;
    const float vi = ((xi + yi) != 0.f) ? 1.f : 0.f;
    const float vj = ((xj + yj) != 0.f) ? 1.f : 0.f;
    sc[e] = d * 0.125f + (1.f / (dist + 1e-8f)) * vi * vj;
  }
  __syncthreads();
  if (lane < 12) {
    float mx = -1e30f;
    for (int j = 0; j < 12; j++) mx = fmaxf(mx, sc[lane * 12 + j]);
    float sm = 0.f;
    for (int j = 0; j < 12; j++) { float e = expf(sc[lane * 12 + j] - mx); sc[lane * 12 + j] = e; sm += e; }
    const float inv = 1.f / sm;
    for (int j = 0; j < 12; j++) sc[lane * 12 + j] *= inv;
  }
  __syncthreads();
  for (int e = lane; e < 768; e += 32) {
    const int i = e / 64, d = e % 64;
    float a = 0.f;
#pragma unroll
    for (int j = 0; j < 12; j++) a += sc[i * 12 + j] * (float)v[j * 768 + d];
    o[(size_t)(s * 12 + i) * 256 + h * 64 + d] = (_Float16)a;
  }
}

// ---- mean over 12 patches ----
__global__ void mean12_k(const float* __restrict__ x, float* __restrict__ xm) {
  const int idx = blockIdx.x * 256 + threadIdx.x;
  if (idx >= SEQ_ * 256) return;
  const int d = idx & 255, s = idx >> 8;
  float acc = 0.f;
#pragma unroll
  for (int i = 0; i < 12; i++) acc += x[((size_t)(s * 12 + i) << 8) + d];
  xm[idx] = acc * (1.f / 12.f);
}

// ---- detached branch: cos-sim graph -> Floyd-Warshall -> Torgerson MDS (Jacobi eigh 6x6) -> pdist ----
__global__ void __launch_bounds__(64) computeM_k(const float* __restrict__ xr,
                                                 float* __restrict__ Mout) {
  const int lane = threadIdx.x & 31, wv = threadIdx.x >> 5;
  const int b = blockIdx.x * 2 + wv;
  const float* xb = xr + (size_t)b * 6 * 256;
  __shared__ float Ash[2][36];
  float* As = Ash[wv];
  float nrm[6];
#pragma unroll
  for (int i = 0; i < 6; i++) {
    float s = 0.f;
    for (int t = lane; t < 256; t += 32) { float u = xb[i * 256 + t]; s += u * u; }
    nrm[i] = fmaxf(sqrtf(wave_sum(s)), 1e-8f);
  }
  for (int i = 0; i < 6; i++)
    for (int j = i; j < 6; j++) {
      float s = 0.f;
      for (int t = lane; t < 256; t += 32) s += xb[i * 256 + t] * xb[j * 256 + t];
      s = wave_sum(s);
      if (lane == 0) {
        const float a = fmaxf(s / (nrm[i] * nrm[j]), 0.1f);
        As[i * 6 + j] = a; As[j * 6 + i] = a;
      }
    }
  __syncthreads();
  if (lane != 0) return;

  float D[36];
  for (int e = 0; e < 36; e++) D[e] = ((e / 6) == (e % 6)) ? 0.f : As[e];
  for (int k = 0; k < 6; k++)
    for (int i = 0; i < 6; i++)
      for (int j = 0; j < 6; j++)
        D[i * 6 + j] = fminf(D[i * 6 + j], D[i * 6 + k] + D[k * 6 + j]);
  float rm[6], cm[6], tm = 0.f;
  for (int e = 0; e < 36; e++) D[e] *= D[e];
  for (int i = 0; i < 6; i++) {
    float s = 0.f;
    for (int j = 0; j < 6; j++) s += D[i * 6 + j];
    rm[i] = s * (1.f / 6.f); tm += s;
  }
  for (int j = 0; j < 6; j++) {
    float s = 0.f;
    for (int i = 0; i < 6; i++) s += D[i * 6 + j];
    cm[j] = s * (1.f / 6.f);
  }
  tm *= (1.f / 36.f);
  float S[36], V[36];
  for (int i = 0; i < 6; i++)
    for (int j = 0; j < 6; j++) {
      S[i * 6 + j] = -0.5f * (D[i * 6 + j] - rm[i] - cm[j] + tm);
      V[i * 6 + j] = (i == j) ? 1.f : 0.f;
    }
  for (int sweep = 0; sweep < 10; sweep++)
    for (int p = 0; p < 5; p++)
      for (int qq = p + 1; qq < 6; qq++) {
        const float apq = S[p * 6 + qq];
        if (fabsf(apq) < 1e-12f) continue;
        const float tau = (S[qq * 6 + qq] - S[p * 6 + p]) / (2.f * apq);
        const float t = ((tau >= 0.f) ? 1.f : -1.f) / (fabsf(tau) + sqrtf(1.f + tau * tau));
        const float c = rsqrtf(1.f + t * t);
        const float s = t * c;
        for (int r = 0; r < 6; r++) {
          const float a = S[r * 6 + p], bq = S[r * 6 + qq];
          S[r * 6 + p] = c * a - s * bq; S[r * 6 + qq] = s * a + c * bq;
        }
        for (int r = 0; r < 6; r++) {
          const float a = S[p * 6 + r], bq = S[qq * 6 + r];
          S[p * 6 + r] = c * a - s * bq; S[qq * 6 + r] = s * a + c * bq;
        }
        for (int r = 0; r < 6; r++) {
          const float a = V[r * 6 + p], bq = V[r * 6 + qq];
          V[r * 6 + p] = c * a - s * bq; V[r * 6 + qq] = s * a + c * bq;
        }
      }
  int i1 = 0;
  for (int k = 1; k < 6; k++) if (S[k * 7] > S[i1 * 7]) i1 = k;
  int i2 = (i1 == 0) ? 1 : 0;
  for (int k = 0; k < 6; k++) if (k != i1 && S[k * 7] > S[i2 * 7]) i2 = k;
  const float s1 = sqrtf(fmaxf(S[i1 * 7], 0.f)), s2 = sqrtf(fmaxf(S[i2 * 7], 0.f));
  float Y[12];
  for (int i = 0; i < 6; i++) { Y[2 * i] = V[i * 6 + i2] * s2; Y[2 * i + 1] = V[i * 6 + i1] * s1; }
  float Md[36];
  for (int i = 0; i < 6; i++)
    for (int j = 0; j < 6; j++) {
      const float dx = Y[2 * i] - Y[2 * j], dy = Y[2 * i + 1] - Y[2 * j + 1];
      const float d2 = dx * dx + dy * dy;
      Md[i * 6 + j] = (d2 > 0.f) ? sqrtf(d2) : 0.f;
    }
  for (int i = 0; i < 6; i++) {
    float mx = 0.f;
    for (int j = 0; j < 6; j++) mx = fmaxf(mx, Md[i * 6 + j]);
    mx = fmaxf(mx, 1e-12f);
    for (int j = 0; j < 6; j++) Mout[b * 36 + i * 6 + j] = Md[i * 6 + j] / mx;
  }
}

// ---- GCN adj multiply + ReLU: y = relu((alpha*M + (1-alpha)*AU) @ t) ----
__global__ void gcn_adj_k(const float* __restrict__ t, const float* __restrict__ Mm,
                          const float* __restrict__ AU, const float* __restrict__ alpha,
                          float* __restrict__ y) {
  const int idx = blockIdx.x * 256 + threadIdx.x;
  if (idx >= B_ * 6 * 256) return;
  const int d = idx & 255, bi = idx >> 8, i = bi % 6, b = bi / 6;
  const float al = alpha[0];
  float acc = 0.f;
#pragma unroll
  for (int j = 0; j < 6; j++) {
    const float a = al * Mm[b * 36 + i * 6 + j] + (1.f - al) * AU[i * 6 + j];
    acc += a * t[((size_t)(b * 6 + j) << 8) + d];
  }
  y[idx] = fmaxf(acc, 0.f);
}

// ---- head: mean over 6, LN, [256,3] projection; one wave per batch ----
__global__ void __launch_bounds__(256) head_k(const float* __restrict__ x,
                                              const float* __restrict__ g,
                                              const float* __restrict__ bl,
                                              const float* __restrict__ wh,
                                              const float* __restrict__ bh,
                                              float* __restrict__ out) {
  const int lane = threadIdx.x & 31, wv = threadIdx.x >> 5;
  const int b = blockIdx.x * 8 + wv;
  if (b >= B_) return;
  const float* xb = x + (size_t)b * 6 * 256;
  float v[8], s = 0.f, ss = 0.f;
#pragma unroll
  for (int i = 0; i < 8; i++) {
    const int d = lane * 8 + i;
    float acc = 0.f;
#pragma unroll
    for (int r = 0; r < 6; r++) acc += xb[r * 256 + d];
    acc *= (1.f / 6.f);
    v[i] = acc; s += acc; ss += acc * acc;
  }
  s = wave_sum(s); ss = wave_sum(ss);
  const float m = s * (1.f / 256.f);
  const float r = rsqrtf(ss * (1.f / 256.f) - m * m + 1e-5f);
  float hv[8];
#pragma unroll
  for (int i = 0; i < 8; i++) { const int d = lane * 8 + i; hv[i] = (v[i] - m) * r * g[d] + bl[d]; }
  for (int c = 0; c < 3; c++) {
    float acc = 0.f;
#pragma unroll
    for (int i = 0; i < 8; i++) acc += hv[i] * wh[(lane * 8 + i) * 3 + c];
    acc = wave_sum(acc);
    if (lane == 0) out[b * 3 + c] = acc + bh[c];
  }
}

// ---- weight prep: transpose + f16 convert; Weff = w_patch @ w_comb_top; beff per region ----
__global__ void cvt_wT_k(const float* __restrict__ W, _Float16* __restrict__ Bt, int K, int N) {
  const int idx = blockIdx.x * 256 + threadIdx.x;
  if (idx >= N * K) return;
  const int n = idx / K, k = idx % K;
  Bt[idx] = (_Float16)W[k * N + n];
}
__global__ void weffT_k(const float* __restrict__ wp, const float* __restrict__ wc,
                        _Float16* __restrict__ WT) {
  const int idx = blockIdx.x * 256 + threadIdx.x;
  if (idx >= 256 * PDP_) return;
  const int d = idx / PDP_, k = idx % PDP_;
  if (k >= PD_) { WT[idx] = (_Float16)0.f; return; }
  float s = 0.f;
  for (int j = 0; j < 256; j++) s += wp[k * 256 + j] * wc[j * 256 + d];
  WT[idx] = (_Float16)s;
}
__global__ void beff_k(const float* __restrict__ bp, const float* __restrict__ wc,
                       const float* __restrict__ emb, const float* __restrict__ bc,
                       float* __restrict__ beff) {
  const int idx = blockIdx.x * 256 + threadIdx.x;
  if (idx >= 6 * 256) return;
  const int r = idx / 256, d = idx % 256;
  float s = bc[d];
  for (int j = 0; j < 256; j++) s += bp[j] * wc[j * 256 + d];
  for (int t = 0; t < 32; t++) s += emb[r * 32 + t] * wc[(256 + t) * 256 + d];
  beff[idx] = s;
}
__global__ void pat_cvt_k(const float* __restrict__ p, _Float16* __restrict__ p16) {
  const int idx = blockIdx.x * 256 + threadIdx.x;
  if (idx >= TOK_ * PDP_) return;
  const int m = idx / PDP_, k = idx % PDP_;
  p16[idx] = (k < PD_) ? (_Float16)p[(size_t)m * PD_ + k] : (_Float16)0.f;
}

extern "C" void kernel_launch(void* const* d_in, const int* in_sizes, int n_in,
                              void* d_out, int out_size, void* d_ws, size_t ws_size,
                              hipStream_t stream) {
  (void)in_sizes; (void)n_in; (void)out_size; (void)ws_size;
  const float* patches   = (const float*)d_in[0];
  const float* keypoints = (const float*)d_in[2];
  const float* w_patch = (const float*)d_in[3];
  const float* b_patch = (const float*)d_in[4];
  const float* emb     = (const float*)d_in[5];
  const float* w_comb  = (const float*)d_in[6];
  const float* b_comb  = (const float*)d_in[7];
  struct Layer { const float *ln1g,*ln1b,*wqkv,*wo,*wob,*ln2g,*ln2b,*w1,*b1,*w2,*b2; };
  Layer L[3];
  for (int l = 0; l < 3; l++) {
    const int o = 8 + 11 * l;
    L[l] = { (const float*)d_in[o+0], (const float*)d_in[o+1], (const float*)d_in[o+2],
             (const float*)d_in[o+3], (const float*)d_in[o+4], (const float*)d_in[o+5],
             (const float*)d_in[o+6], (const float*)d_in[o+7], (const float*)d_in[o+8],
             (const float*)d_in[o+9], (const float*)d_in[o+10] };
  }
  const float* agg_g  = (const float*)d_in[41];
  const float* agg_b  = (const float*)d_in[42];
  const float* w_agg  = (const float*)d_in[43];
  const float* b_agg  = (const float*)d_in[44];
  const float* gcn_w[3] = { (const float*)d_in[45], (const float*)d_in[47], (const float*)d_in[49] };
  const float* gcn_b[3] = { (const float*)d_in[46], (const float*)d_in[48], (const float*)d_in[50] };
  const float* head_g = (const float*)d_in[51];
  const float* head_b = (const float*)d_in[52];
  const float* w_head = (const float*)d_in[53];
  const float* b_head = (const float*)d_in[54];
  const float* alpha  = (const float*)d_in[55];
  const float* AU     = (const float*)d_in[56];
  float* out = (float*)d_out;

  char* ws = (char*)d_ws;
  size_t cur = 0;
  auto alloc = [&](size_t bytes) -> char* {
    char* p = ws + cur;
    cur = (cur + bytes + 255) & ~(size_t)255;
    return p;
  };
  _Float16* weffT = (_Float16*)alloc((size_t)256 * PDP_ * 2);
  float*    beff  = (float*)alloc(6 * 256 * 4);
  _Float16 *wqkvT[3], *woT[3], *w1T[3], *w2T[3], *gcnT[3];
  for (int l = 0; l < 3; l++) {
    wqkvT[l] = (_Float16*)alloc((size_t)768 * 256 * 2);
    woT[l]   = (_Float16*)alloc((size_t)256 * 256 * 2);
    w1T[l]   = (_Float16*)alloc((size_t)1024 * 256 * 2);
    w2T[l]   = (_Float16*)alloc((size_t)256 * 1024 * 2);
    gcnT[l]  = (_Float16*)alloc((size_t)256 * 256 * 2);
  }
  _Float16* waggT = (_Float16*)alloc((size_t)256 * 256 * 2);
  _Float16* pat16 = (_Float16*)alloc((size_t)TOK_ * PDP_ * 2);
  float*    x     = (float*)alloc((size_t)TOK_ * 256 * 4);
  _Float16* h16   = (_Float16*)alloc((size_t)TOK_ * 256 * 2);   // LN out, also attention O
  _Float16* big16 = (_Float16*)alloc((size_t)TOK_ * 1024 * 2);  // qkv (ld=768) / mlp hidden (ld=1024)
  float*    xm    = (float*)alloc((size_t)SEQ_ * 256 * 4);
  float*    xr    = (float*)alloc((size_t)SEQ_ * 256 * 4);
  float*    xg    = (float*)alloc((size_t)SEQ_ * 256 * 4);
  float*    tg    = (float*)alloc((size_t)SEQ_ * 256 * 4);
  float*    Mmat  = (float*)alloc((size_t)B_ * 36 * 4);

  const dim3 blk(256);
  // weight prep (cheap, rerun each call for determinism)
  weffT_k<<<(256 * PDP_ + 255) / 256, blk, 0, stream>>>(w_patch, w_comb, weffT);
  beff_k<<<6, blk, 0, stream>>>(b_patch, w_comb, emb, b_comb, beff);
  for (int l = 0; l < 3; l++) {
    cvt_wT_k<<<(768 * 256 + 255) / 256, blk, 0, stream>>>(L[l].wqkv, wqkvT[l], 256, 768);
    cvt_wT_k<<<(256 * 256 + 255) / 256, blk, 0, stream>>>(L[l].wo, woT[l], 256, 256);
    cvt_wT_k<<<(1024 * 256 + 255) / 256, blk, 0, stream>>>(L[l].w1, w1T[l], 256, 1024);
    cvt_wT_k<<<(256 * 1024 + 255) / 256, blk, 0, stream>>>(L[l].w2, w2T[l], 1024, 256);
    cvt_wT_k<<<(256 * 256 + 255) / 256, blk, 0, stream>>>(gcn_w[l], gcnT[l], 256, 256);
  }
  cvt_wT_k<<<(256 * 256 + 255) / 256, blk, 0, stream>>>(w_agg, waggT, 256, 256);
  pat_cvt_k<<<(TOK_ * PDP_ + 255) / 256, blk, 0, stream>>>(patches, pat16);

  // fused patch+region embed: x = pat16 @ weffT^T + beff[region]
  gemm_k<0, 3><<<dim3(256 / 64, TOK_ / 256), blk, 0, stream>>>(pat16, weffT, beff, nullptr, x,
                                                               TOK_, 256, PDP_);
  // 3 encoder layers
  for (int l = 0; l < 3; l++) {
    ln_k<<<TOK_ / 8, blk, 0, stream>>>(x, L[l].ln1g, L[l].ln1b, h16, TOK_);
    gemm_k<0, 0><<<dim3(768 / 64, TOK_ / 256), blk, 0, stream>>>(h16, wqkvT[l], nullptr, nullptr,
                                                                 big16, TOK_, 768, 256);
    attn_k<<<(SEQ_ * 4) / 8, blk, 0, stream>>>(big16, keypoints, h16);
    gemm_k<0, 2><<<dim3(256 / 64, TOK_ / 256), blk, 0, stream>>>(h16, woT[l], L[l].wob, x, x,
                                                                 TOK_, 256, 256);
    ln_k<<<TOK_ / 8, blk, 0, stream>>>(x, L[l].ln2g, L[l].ln2b, h16, TOK_);
    gemm_k<0, 1><<<dim3(1024 / 64, TOK_ / 256), blk, 0, stream>>>(h16, w1T[l], L[l].b1, nullptr,
                                                                  big16, TOK_, 1024, 256);
    gemm_k<0, 2><<<dim3(256 / 64, TOK_ / 256), blk, 0, stream>>>(big16, w2T[l], L[l].b2, x, x,
                                                                 TOK_, 256, 1024);
  }
  // aggregate per region
  mean12_k<<<(SEQ_ * 256 + 255) / 256, blk, 0, stream>>>(x, xm);
  ln_k<<<SEQ_ / 8, blk, 0, stream>>>(xm, agg_g, agg_b, h16, SEQ_);
  gemm_k<0, 2><<<dim3(256 / 64, SEQ_ / 256), blk, 0, stream>>>(h16, waggT, b_agg, nullptr, xr,
                                                               SEQ_, 256, 256);
  // detached MDS branch
  computeM_k<<<B_ / 2, dim3(64), 0, stream>>>(xr, Mmat);
  // GCN: xr -> xg -> xr -> xg
  float* cu = xr;
  float* nx = xg;
  for (int l = 0; l < 3; l++) {
    gemm_k<1, 2><<<dim3(256 / 64, SEQ_ / 256), blk, 0, stream>>>(cu, gcnT[l], gcn_b[l], nullptr,
                                                                 tg, SEQ_, 256, 256);
    gcn_adj_k<<<(B_ * 6 * 256 + 255) / 256, blk, 0, stream>>>(tg, Mmat, AU, alpha, nx);
    float* tmp = cu; cu = nx; nx = tmp;
  }
  // head
  head_k<<<B_ / 8, blk, 0, stream>>>(cu, head_g, head_b, w_head, b_head, out);
}
